// HeteroGNN_29394756174084
// MI455X (gfx1250) — matrix operand order repaired and checked by reading, not compiled, not run on hardware
//
#include <hip/hip_runtime.h>
#include <hip/hip_bf16.h>

typedef __attribute__((ext_vector_type(16))) _Float16 v16h;
typedef __attribute__((ext_vector_type(8)))  float    v8f;
typedef _Float16 half_t;

#define NN 50000
#define EE 800000
#define DD 128
#define GG 16
#define LL 3
#define PER 16384   // halves per packed 128x128 matrix (8 ntiles * 4 kchunks * 32 lanes * 16)

// ---------------------------------------------------------------- utilities

__global__ void zero_k(float* __restrict__ p, size_t n) {
    size_t i = (size_t)blockIdx.x * blockDim.x + threadIdx.x;
    size_t stride = (size_t)gridDim.x * blockDim.x;
    for (; i < n; i += stride) p[i] = 0.0f;
}

// degree count: cnt[dst[e]] += 1
__global__ __launch_bounds__(256) void degree_k(const int* __restrict__ dst,
                                                float* __restrict__ cnt, int e) {
    int i = blockIdx.x * blockDim.x + threadIdx.x;
    if (i < e) atomicAdd(&cnt[dst[i]], 1.0f);
}

// ---------------------------------------------------------------- weight prep
// Split every 128x128 weight into f16 hi + f16 lo, packed in B-fragment order:
// packed[((nt*4 + kcI)*32 + lane)*16 + e] = W[kcI*32 + (e<8 ? 8*(lane>>4)+e : 16+8*(lane>>4)+(e-8))][nt*16 + (lane&15)]
__global__ __launch_bounds__(256) void prep_k(const float* __restrict__ Wl,
                                              const float* __restrict__ Wr,
                                              half_t* __restrict__ wpack) {
    int idx = blockIdx.x * blockDim.x + threadIdx.x;
    if (idx >= 6 * 2 * PER) return;
    int lt    = idx / (2 * PER);        // layer*2 + type  (0..5)
    int rem   = idx % (2 * PER);
    int which = rem / PER;              // 0 = Wl, 1 = Wr
    int pos   = rem % PER;
    int e    = pos & 15;
    int lane = (pos >> 4) & 31;
    int kcI  = (pos >> 9) & 3;
    int nt   = pos >> 11;
    int n = nt * 16 + (lane & 15);
    int k = kcI * 32 + (e < 8 ? 8 * (lane >> 4) + e : 16 + 8 * (lane >> 4) + (e - 8));
    const float* W = which ? Wr : Wl;
    float w = W[((size_t)lt * DD + k) * DD + n];
    _Float16 hi = (_Float16)w;
    _Float16 lo = (_Float16)(w - (float)hi);
    half_t* dstb = wpack + (size_t)lt * 4 * PER + (size_t)which * 2 * PER;
    dstb[pos]       = hi;
    dstb[PER + pos] = lo;
}

// ---------------------------------------------------------------- edge scatter
// agg[dst] += x[src]; one wave (32 lanes) per edge, float4 per lane.
__global__ __launch_bounds__(256) void scatter_k(const float* __restrict__ X,
                                                 const int* __restrict__ src,
                                                 const int* __restrict__ dst,
                                                 float* __restrict__ agg, int E) {
    int e = blockIdx.x * 8 + (threadIdx.x >> 5);
    int lane = threadIdx.x & 31;
    if (e >= E) return;
    int s = src[e], d = dst[e];
    const float4 v = ((const float4*)(X + (size_t)s * DD))[lane];
    float* ad = agg + (size_t)d * DD + lane * 4;
    atomicAdd(ad + 0, v.x);
    atomicAdd(ad + 1, v.y);
    atomicAdd(ad + 2, v.z);
    atomicAdd(ad + 3, v.w);
}

// ---------------------------------------------------------------- SAGE layer GEMM
// out = relu( (agg/max(cnt,1)) @ Wl + x @ Wr + b )   via split-f16 WMMA.
__global__ __launch_bounds__(128) void sage_gemm_k(
        const float* __restrict__ agg, const float* __restrict__ cnt,
        const float* __restrict__ xin, const half_t* __restrict__ wp,
        const float* __restrict__ bias, float* __restrict__ xout) {
    const half_t* WlHi = wp;
    const half_t* WlLo = wp + PER;
    const half_t* WrHi = wp + 2 * PER;
    const half_t* WrLo = wp + 3 * PER;

    int tile = blockIdx.x;            // 16 node rows per block; 50000/16 = 3125 exact
    int wv   = threadIdx.x >> 5;      // wave 0..3
    int lane = threadIdx.x & 31;
    int m = lane & 15;                // A: row index ; C: column-within-tile
    int h = lane >> 4;                // lane-half selector
    int row = tile * 16 + m;

    const float* arow = agg + (size_t)row * DD;
    const float* xrow = xin + (size_t)row * DD;
    float invc = 1.0f / fmaxf(cnt[row], 1.0f);

    for (int s = 0; s < 2; ++s) {
        int nt = wv + 4 * s;          // N-tile 0..7 (16 cols each)
        v8f acc = {};
        #pragma unroll
        for (int kcI = 0; kcI < 4; ++kcI) {
            int kc = kcI * 32;
            // --- A fragments (16-bit A 16x32 layout): lane m holds row m,
            //     e=0..7 -> K = kc+8h+e ; e=8..15 -> K = kc+16+8h+(e-8)
            float av0[8], av1[8], xv0[8], xv1[8];
            {
                float4 a = ((const float4*)(arow + kc + 8 * h))[0];
                float4 b = ((const float4*)(arow + kc + 8 * h))[1];
                av0[0]=a.x; av0[1]=a.y; av0[2]=a.z; av0[3]=a.w;
                av0[4]=b.x; av0[5]=b.y; av0[6]=b.z; av0[7]=b.w;
                float4 c = ((const float4*)(arow + kc + 16 + 8 * h))[0];
                float4 d = ((const float4*)(arow + kc + 16 + 8 * h))[1];
                av1[0]=c.x; av1[1]=c.y; av1[2]=c.z; av1[3]=c.w;
                av1[4]=d.x; av1[5]=d.y; av1[6]=d.z; av1[7]=d.w;
                float4 e = ((const float4*)(xrow + kc + 8 * h))[0];
                float4 f = ((const float4*)(xrow + kc + 8 * h))[1];
                xv0[0]=e.x; xv0[1]=e.y; xv0[2]=e.z; xv0[3]=e.w;
                xv0[4]=f.x; xv0[5]=f.y; xv0[6]=f.z; xv0[7]=f.w;
                float4 g = ((const float4*)(xrow + kc + 16 + 8 * h))[0];
                float4 q = ((const float4*)(xrow + kc + 16 + 8 * h))[1];
                xv1[0]=g.x; xv1[1]=g.y; xv1[2]=g.z; xv1[3]=g.w;
                xv1[4]=q.x; xv1[5]=q.y; xv1[6]=q.z; xv1[7]=q.w;
            }
            v16h mh, ml, xh, xl;
            #pragma unroll
            for (int j = 0; j < 8; ++j) {
                float va = av0[j] * invc, vb = av1[j] * invc;
                _Float16 ha = (_Float16)va, hb = (_Float16)vb;
                mh[j] = ha; mh[8 + j] = hb;
                ml[j]     = (_Float16)(va - (float)ha);
                ml[8 + j] = (_Float16)(vb - (float)hb);
                float vx = xv0[j], vy = xv1[j];
                _Float16 hx = (_Float16)vx, hy = (_Float16)vy;
                xh[j] = hx; xh[8 + j] = hy;
                xl[j]     = (_Float16)(vx - (float)hx);
                xl[8 + j] = (_Float16)(vy - (float)hy);
            }
            // --- B fragments: prepacked, one contiguous 32B load per fragment
            size_t boff = (size_t)(((nt * 4 + kcI) * 32 + lane)) << 4;
            v16h blh = *(const v16h*)(WlHi + boff);
            v16h bll = *(const v16h*)(WlLo + boff);
            v16h brh = *(const v16h*)(WrHi + boff);
            v16h brl = *(const v16h*)(WrLo + boff);
            // --- split-f16 fp32-accurate MMA: hi*hi + lo*hi + hi*lo
            acc = __builtin_amdgcn_wmma_f32_16x16x32_f16(false, mh, false, blh, (short)0, acc, false, false);
            acc = __builtin_amdgcn_wmma_f32_16x16x32_f16(false, ml, false, blh, (short)0, acc, false, false);
            acc = __builtin_amdgcn_wmma_f32_16x16x32_f16(false, mh, false, bll, (short)0, acc, false, false);
            acc = __builtin_amdgcn_wmma_f32_16x16x32_f16(false, xh, false, brh, (short)0, acc, false, false);
            acc = __builtin_amdgcn_wmma_f32_16x16x32_f16(false, xl, false, brh, (short)0, acc, false, false);
            acc = __builtin_amdgcn_wmma_f32_16x16x32_f16(false, xh, false, brl, (short)0, acc, false, false);
        }
        // --- epilogue: C layout — lane&15 is column n, VGPR r -> row r + 8h
        int n_col = nt * 16 + m;
        float b = bias[n_col];
        #pragma unroll
        for (int r = 0; r < 8; ++r) {
            int mr = r + 8 * h;
            float v = fmaxf(acc[r] + b, 0.0f);
            xout[(size_t)(tile * 16 + mr) * DD + n_col] = v;
        }
    }
}

// ---------------------------------------------------------------- graph mean-pool
// LDS-staged per-graph sums (ds_add_f32), then one global atomic per LDS slot.
__global__ __launch_bounds__(256) void pool_k(const float* __restrict__ X,
                                              const int* __restrict__ batch,
                                              float* __restrict__ pool,
                                              float* __restrict__ cntg, int n) {
    __shared__ float lp[GG * DD];
    __shared__ float lc[GG];
    for (int i = threadIdx.x; i < GG * DD; i += 256) lp[i] = 0.0f;
    if (threadIdx.x < GG) lc[threadIdx.x] = 0.0f;
    __syncthreads();
    int wv = threadIdx.x >> 5, lane = threadIdx.x & 31;
    int base = blockIdx.x * 256;
    for (int node = base + wv; node < base + 256; node += 8) {
        if (node < n) {
            int g = batch[node];
            const float4 v = ((const float4*)(X + (size_t)node * DD))[lane];
            float* d = lp + g * DD + lane * 4;
            atomicAdd(d + 0, v.x);
            atomicAdd(d + 1, v.y);
            atomicAdd(d + 2, v.z);
            atomicAdd(d + 3, v.w);
            if (lane == 0) atomicAdd(&lc[g], 1.0f);
        }
    }
    __syncthreads();
    for (int i = threadIdx.x; i < GG * DD; i += 256) atomicAdd(&pool[i], lp[i]);
    if (threadIdx.x < GG) atomicAdd(&cntg[threadIdx.x], lc[threadIdx.x]);
}

// ---------------------------------------------------------------- final linear
// out[g,o] = sum_k concat(pool_void/c, pool_vessel/c)[g,k] * lin_W[k,o] + lin_b[o]
__global__ void final_k(const float* __restrict__ pool, const float* __restrict__ cntg,
                        const float* __restrict__ linW, const float* __restrict__ linb,
                        float* __restrict__ out) {
    int i = threadIdx.x;          // 0..31 = G*OUT
    if (i >= GG * 2) return;
    int g = i >> 1, o = i & 1;
    float s = linb[o];
    for (int t = 0; t < 2; ++t) {
        float ic = 1.0f / fmaxf(cntg[t * GG + g], 1.0f);
        const float* pg = pool + (size_t)(t * GG + g) * DD;
        for (int k = 0; k < DD; ++k)
            s += pg[k] * ic * linW[(t * DD + k) * 2 + o];
    }
    out[g * 2 + o] = s;
}

// ---------------------------------------------------------------- launch

extern "C" void kernel_launch(void* const* d_in, const int* in_sizes, int n_in,
                              void* d_out, int out_size, void* d_ws, size_t ws_size,
                              hipStream_t stream) {
    const float* x0   = (const float*)d_in[0];
    const float* x1   = (const float*)d_in[1];
    const float* Wl   = (const float*)d_in[2];
    const float* bl   = (const float*)d_in[3];
    const float* Wr   = (const float*)d_in[4];
    const float* linW = (const float*)d_in[5];
    const float* linb = (const float*)d_in[6];
    const int* edge[2]  = {(const int*)d_in[7], (const int*)d_in[8]};
    const int* batch[2] = {(const int*)d_in[9], (const int*)d_in[10]};
    float* out = (float*)d_out;

    // ---- carve workspace (~154 MB)
    char* p = (char*)d_ws;
    auto carve = [&](size_t bytes) -> char* {
        char* r = p;
        p += (bytes + 255) & ~(size_t)255;
        return r;
    };
    float* cnt    = (float*)carve((size_t)2 * NN * sizeof(float));     // degrees, both types
    float* agg[2] = {(float*)carve((size_t)NN * DD * sizeof(float)),
                     (float*)carve((size_t)NN * DD * sizeof(float))};
    float* xb[2][2];
    for (int t = 0; t < 2; ++t)
        for (int i = 0; i < 2; ++i)
            xb[t][i] = (float*)carve((size_t)NN * DD * sizeof(float)); // ping-pong features
    half_t* wpack = (half_t*)carve((size_t)6 * 4 * PER * sizeof(half_t));
    float* pool   = (float*)carve((size_t)(2 * GG * DD + 2 * GG) * sizeof(float));
    float* cntg   = pool + 2 * GG * DD;

    // ---- per-launch init (no cross-call state)
    zero_k<<<512, 256, 0, stream>>>(cnt, (size_t)2 * NN);
    zero_k<<<16, 256, 0, stream>>>(pool, (size_t)(2 * GG * DD + 2 * GG));
    degree_k<<<(EE + 255) / 256, 256, 0, stream>>>(edge[0] + EE, cnt, EE);
    degree_k<<<(EE + 255) / 256, 256, 0, stream>>>(edge[1] + EE, cnt + NN, EE);
    prep_k<<<(6 * 2 * PER + 255) / 256, 256, 0, stream>>>(Wl, Wr, wpack);

    // ---- 3 SAGE layers, 2 node types
    const float* xin[2] = {x0, x1};
    for (int l = 0; l < LL; ++l) {
        for (int t = 0; t < 2; ++t) {
            zero_k<<<2048, 256, 0, stream>>>(agg[t], (size_t)NN * DD);
            scatter_k<<<(EE + 7) / 8, 256, 0, stream>>>(xin[t], edge[t], edge[t] + EE,
                                                        agg[t], EE);
            sage_gemm_k<<<NN / 16, 128, 0, stream>>>(
                agg[t], cnt + (size_t)t * NN, xin[t],
                wpack + (size_t)(l * 2 + t) * 4 * PER,
                bl + (size_t)(l * 2 + t) * DD,
                xb[t][l & 1]);
            xin[t] = xb[t][l & 1];
        }
    }

    // ---- pooling + classifier head
    pool_k<<<(NN + 255) / 256, 256, 0, stream>>>(xin[0], batch[0], pool, cntg, NN);
    pool_k<<<(NN + 255) / 256, 256, 0, stream>>>(xin[1], batch[1], pool + GG * DD,
                                                 cntg + GG, NN);
    final_k<<<1, 32, 0, stream>>>(pool, cntg, linW, linb, out);
}
